// Model_39676907882392
// MI455X (gfx1250) — matrix operand order repaired and checked by reading, not compiled, not run on hardware
//
#include <hip/hip_runtime.h>

typedef __attribute__((ext_vector_type(16))) _Float16 v16h;
typedef __attribute__((ext_vector_type(8)))  _Float16 v8h;
typedef __attribute__((ext_vector_type(8)))  float    v8f;

#define BATCH 4
#define SEQ   4096
#define DM    1024
#define NROWS (BATCH * SEQ)

__device__ __forceinline__ v8f wmma_f16(v16h a, v16h b, v8f c) {
    return __builtin_amdgcn_wmma_f32_16x16x32_f16(
        /*neg_a=*/false, a, /*neg_b=*/false, b,
        /*c_mod=*/(short)0, c, /*reuse_a=*/false, /*reuse_b=*/false);
}

__device__ __forceinline__ v16h load_frag16(const _Float16* p) {
    v8h x0 = *(const v8h*)p;
    v8h x1 = *(const v8h*)(p + 8);
    v16h r;
#pragma unroll
    for (int t = 0; t < 8; ++t) { r[t] = x0[t]; r[8 + t] = x1[t]; }
    return r;
}

// ---------------------------------------------------------------------------
// Projection: Y[r, e] = sum_d X[r, d] * W[e, d] + b0[e] (+ b1[e]); f16 output.
// Register-blocked: one wave computes a 16x64 output tile (one A-fragment
// reused across 4 B-fragments -> 4 WMMA per K-step of 32). Loads for all
// fragments are issued before the WMMA chain to hide latency.
// HAS_YT also writes transposed copy Yt[b][e][s] (b = r/SEQ, s = r%SEQ).
// ---------------------------------------------------------------------------
template <bool HAS_B1, bool HAS_YT>
__global__ __launch_bounds__(256) void proj_kernel(
    const float* __restrict__ X,    // [NROWS, DM] fp32
    const float* __restrict__ W,    // [DM, DM]  fp32 (row e, col d)
    const float* __restrict__ b0,   // [DM]
    const float* __restrict__ b1,   // [DM]
    _Float16* __restrict__ Y,       // [NROWS, DM] f16
    _Float16* __restrict__ Yt)      // [BATCH, DM, SEQ] f16
{
    const int lane = threadIdx.x & 31;
    const int wave = threadIdx.x >> 5;
    const int half = lane >> 4;   // 0: lanes 0-15, 1: lanes 16-31
    const int l16  = lane & 15;

    const int tilesPerRow = DM / 64;                      // 16 col-tiles of 64
    const int tile = blockIdx.x * 8 + wave;
    const int tm = tile / tilesPerRow;                    // 16-row tile
    const int tn = tile % tilesPerRow;                    // 64-col tile

    const int arow = tm * 16 + l16;   // A-matrix row held by this lane

    v8f acc[4];
#pragma unroll
    for (int nt = 0; nt < 4; ++nt)
        acc[nt] = (v8f){0.f,0.f,0.f,0.f,0.f,0.f,0.f,0.f};

    for (int k = 0; k < DM; k += 32) {
        // ---- issue all fp32 loads first (A + 4 B tiles), then convert+wmma ----
        // A fragment (16x32 f16): lane half h holds K = {16h..16h+15}
        const float* ap = X + (size_t)arow * DM + k + half * 16;
        float af[16];
#pragma unroll
        for (int t = 0; t < 16; ++t) { af[t] = ap[t]; }

        float bf[4][16];
#pragma unroll
        for (int nt = 0; nt < 4; ++nt) {
            // B fragment (32x16 f16): lane N=l16 -> W row e; K contiguous in d
            const int brow = tn * 64 + nt * 16 + l16;
            const float* bp = W + (size_t)brow * DM + k + half * 16;
#pragma unroll
            for (int t = 0; t < 16; ++t) bf[nt][t] = bp[t];
        }

        v16h a;
#pragma unroll
        for (int t = 0; t < 16; ++t) a[t] = (_Float16)af[t];
#pragma unroll
        for (int nt = 0; nt < 4; ++nt) {
            v16h b;
#pragma unroll
            for (int t = 0; t < 16; ++t) b[t] = (_Float16)bf[nt][t];
            acc[nt] = wmma_f16(a, b, acc[nt]);
        }
    }

    // C layout: VGPR i, lanes 0-15 -> M=i; lanes 16-31 -> M=8+i; N = l16
#pragma unroll
    for (int nt = 0; nt < 4; ++nt) {
        const int col = tn * 64 + nt * 16 + l16;
        float bias = b0[col];
        if (HAS_B1) bias += b1[col];
#pragma unroll
        for (int i = 0; i < 8; ++i) {
            int row = tm * 16 + half * 8 + i;
            float v = acc[nt][i] + bias;
            Y[(size_t)row * DM + col] = (_Float16)v;
            if (HAS_YT) {
                int bb = row >> 12;           // row / SEQ
                int ss = row & (SEQ - 1);     // row % SEQ
                Yt[((size_t)bb * DM + col) * SEQ + ss] = (_Float16)v;
            }
        }
    }
}

// ---------------------------------------------------------------------------
// Fused attention: one workgroup per (batch, 16-row q tile); 8 waves, each
// owning a 128-wide d-slice of the output. Online softmax over 32-key steps.
// Scores never touch HBM; O/m/l live in VGPRs.
// ---------------------------------------------------------------------------
__global__ __launch_bounds__(256) void flash_kernel(
    const _Float16* __restrict__ Qf,   // [B, S, D] f16
    const _Float16* __restrict__ Vf,   // [B, S, D] f16
    const _Float16* __restrict__ Vt,   // [B, D, S] f16
    float* __restrict__ Out)           // [B, S, D] fp32
{
    __shared__ __align__(16) float    sPart[8 * 512];   // per-wave partial scores (16KB)
    __shared__ __align__(16) _Float16 sP[16 * 32];      // P tile, row-major

    const int lane = threadIdx.x & 31;
    const int wave = threadIdx.x >> 5;
    const int half = lane >> 4;
    const int l16  = lane & 15;

    const int b  = blockIdx.x / (SEQ / 16);
    const int qt = blockIdx.x % (SEQ / 16);
    const int q0 = qt * 16;
    const int dw = wave * 128;           // this wave's d-slice [dw, dw+128)

    const _Float16* Qb  = Qf + (size_t)b * SEQ * DM;
    const _Float16* Vb  = Vf + (size_t)b * SEQ * DM;
    const _Float16* Vtb = Vt + (size_t)b * DM * SEQ;

    // Cache Q A-fragments for the wave's 128-wide d-slice (4 chunks of K=32)
    v16h aq[4];
#pragma unroll
    for (int c = 0; c < 4; ++c)
        aq[c] = load_frag16(Qb + (size_t)(q0 + l16) * DM + dw + c * 32 + half * 8 + 8);
    // NOTE: A-fragment uses halves {base..base+15}, base = 16h, the same K order as B;
    // the loads above are overwritten by this explicit build:
#pragma unroll
    for (int c = 0; c < 4; ++c) {
        const _Float16* ap = Qb + (size_t)(q0 + l16) * DM + dw + c * 32 + half * 16;
        v8h lo = *(const v8h*)ap;
        v8h hi = *(const v8h*)(ap + 8);
#pragma unroll
        for (int t = 0; t < 8; ++t) { aq[c][t] = lo[t]; aq[c][8 + t] = hi[t]; }
    }

    v8f o[8];
#pragma unroll
    for (int t = 0; t < 8; ++t) o[t] = (v8f){0.f,0.f,0.f,0.f,0.f,0.f,0.f,0.f};
    float m[8], l[8];
#pragma unroll
    for (int i = 0; i < 8; ++i) { m[i] = -1e30f; l[i] = 0.f; }

    const float sscale = 0.125f;   // 1/sqrt(DM/NUM_HEADS) = 1/8

    for (int k0 = 0; k0 < SEQ; k0 += 32) {
        // ---- phase 1: partial scores S(16q x 32k) over this wave's d-slice ----
        // Load ALL 8 B-fragments first (one clause of b128 loads), then run the
        // WMMA chain so waits are staggered instead of loadcnt==0 per wmma.
        v16h bv[8];
#pragma unroll
        for (int c = 0; c < 4; ++c) {
            // B[k=d][n=key] = Vf[k0+n][d]; lane reads 16 contiguous d at its key row
            bv[2 * c]     = load_frag16(Vb + (size_t)(k0 + l16) * DM + dw + c * 32 + half * 16);
            bv[2 * c + 1] = load_frag16(Vb + (size_t)(k0 + 16 + l16) * DM + dw + c * 32 + half * 16);
        }

        v8f s0 = {0.f,0.f,0.f,0.f,0.f,0.f,0.f,0.f};
        v8f s1 = s0;
#pragma unroll
        for (int c = 0; c < 4; ++c) {
            s0 = wmma_f16(aq[c], bv[2 * c],     s0);
            s1 = wmma_f16(aq[c], bv[2 * c + 1], s1);
        }

        // prefetch next key tile (global_prefetch_b8): QK-side rows + PV-side Vt
        if (k0 + 32 < SEQ) {
            __builtin_prefetch(Vb + (size_t)(k0 + 32 + l16) * DM + dw, 0, 1);
            __builtin_prefetch(Vtb + (size_t)(dw + l16) * SEQ + k0 + 32, 0, 1);
        }

        // stash partials, reduce across the 8 waves through LDS
        {
            float* sp = sPart + wave * 512;
#pragma unroll
            for (int i = 0; i < 8; ++i) {
                sp[lane * 16 + i]     = s0[i];
                sp[lane * 16 + 8 + i] = s1[i];
            }
        }
        __syncthreads();
#pragma unroll
        for (int i = 0; i < 8; ++i) {
            float a0 = 0.f, a1 = 0.f;
#pragma unroll
            for (int w = 0; w < 8; ++w) {
                a0 += sPart[w * 512 + lane * 16 + i];
                a1 += sPart[w * 512 + lane * 16 + 8 + i];
            }
            s0[i] = a0 * sscale;
            s1[i] = a1 * sscale;
        }

        // ---- online softmax (redundant per wave; rows = half*8 + i) ----
        float alpha[8];
#pragma unroll
        for (int i = 0; i < 8; ++i) {
            float rm = fmaxf(s0[i], s1[i]);
#pragma unroll
            for (int msk = 1; msk < 16; msk <<= 1)
                rm = fmaxf(rm, __shfl_xor(rm, msk, 32));
            float mn = fmaxf(m[i], rm);
            alpha[i] = __expf(m[i] - mn);
            float p0 = __expf(s0[i] - mn);
            float p1 = __expf(s1[i] - mn);
            float rs = p0 + p1;
#pragma unroll
            for (int msk = 1; msk < 16; msk <<= 1)
                rs += __shfl_xor(rs, msk, 32);
            l[i] = l[i] * alpha[i] + rs;
            m[i] = mn;
            s0[i] = p0; s1[i] = p1;
        }
#pragma unroll
        for (int t = 0; t < 8; ++t)
#pragma unroll
            for (int i = 0; i < 8; ++i) o[t][i] *= alpha[i];

        // P (16x32) to LDS row-major; identical in all waves -> wave 0 writes
        if (wave == 0) {
#pragma unroll
            for (int i = 0; i < 8; ++i) {
                int row = half * 8 + i;
                sP[row * 32 + l16]      = (_Float16)s0[i];
                sP[row * 32 + 16 + l16] = (_Float16)s1[i];
            }
        }
        __syncthreads();

        // reload P as A-fragment: lane row=l16, K = {16h..16h+15}
        v16h pa;
        {
            const _Float16* pp = sP + l16 * 32 + half * 16;
            v8h x0 = *(const v8h*)pp;
            v8h x1 = *(const v8h*)(pp + 8);
#pragma unroll
            for (int t = 0; t < 8; ++t) { pa[t] = x0[t]; pa[8 + t] = x1[t]; }
        }

        // ---- phase 2: O(16 x 128) += P(16x32) @ V(32 x 128) via Vt ----
        // Same load-all-then-compute structure.
        v16h bw[8];
#pragma unroll
        for (int nt = 0; nt < 8; ++nt) {
            const int dcol = dw + nt * 16 + l16;
            // B[k=key][n=d]: lane N=l16 -> column dcol; K contiguous along S in Vt
            bw[nt] = load_frag16(Vtb + (size_t)dcol * SEQ + k0 + half * 16);
        }
#pragma unroll
        for (int nt = 0; nt < 8; ++nt)
            o[nt] = wmma_f16(pa, bw[nt], o[nt]);
    }

    // ---- epilogue: normalize, store fp32 ----
#pragma unroll
    for (int t = 0; t < 8; ++t) {
        const int dcol = dw + t * 16 + l16;
#pragma unroll
        for (int i = 0; i < 8; ++i) {
            int row = q0 + half * 8 + i;
            Out[((size_t)b * SEQ + row) * DM + dcol] = o[t][i] / l[i];
        }
    }
}

extern "C" void kernel_launch(void* const* d_in, const int* in_sizes, int n_in,
                              void* d_out, int out_size, void* d_ws, size_t ws_size,
                              hipStream_t stream) {
    (void)in_sizes; (void)n_in; (void)out_size;
    const float* query = (const float*)d_in[0];
    const float* value = (const float*)d_in[1];
    const float* Wq    = (const float*)d_in[2];
    const float* bq    = (const float*)d_in[3];
    const float* qk_b  = (const float*)d_in[4];
    const float* Wv    = (const float*)d_in[5];
    const float* bv    = (const float*)d_in[6];
    float* out = (float*)d_out;

    // Workspace partition (f16): Qf [B,S,D], Vf [B,S,D], Vt [B,D,S]
    const size_t elems = (size_t)NROWS * DM;          // 16M halves = 32MB each
    if (ws_size < 3 * elems * sizeof(_Float16)) return;
    _Float16* Qf = (_Float16*)d_ws;
    _Float16* Vf = Qf + elems;
    _Float16* Vt = Vf + elems;

    // Projections: (NROWS/16)*(DM/64) = 16384 wave-tiles, 8 waves per block
    const int tiles  = (NROWS / 16) * (DM / 64);
    const int blocks = tiles / 8;
    proj_kernel<true,  false><<<blocks, 256, 0, stream>>>(query, Wq, bq, qk_b, Qf, nullptr);
    proj_kernel<false, true ><<<blocks, 256, 0, stream>>>(value, Wv, bv, nullptr, Vf, Vt);

    // Fused attention: one block per (batch, 16-row q tile)
    flash_kernel<<<BATCH * (SEQ / 16), 256, 0, stream>>>(Qf, Vf, Vt, out);
}
